// Block_55448027791422
// MI455X (gfx1250) — matrix operand order repaired and checked by reading, not compiled, not run on hardware
//
#include <hip/hip_runtime.h>
#include <math.h>

// ---------------------------------------------------------------------------
// BEiT-style transformer block for MI455X (gfx1250, wave32, WMMA bf16,
// async global->LDS staging, double-buffered GEMM)
// ---------------------------------------------------------------------------

#define DEV __device__ __forceinline__

typedef __attribute__((ext_vector_type(16))) __bf16 bf16x16;
typedef __attribute__((ext_vector_type(8)))  float  f32x8;

union Frag {
    bf16x16        v;
    unsigned int   d[8];
    unsigned short u[16];
};

// ---- dimensions -----------------------------------------------------------
constexpr int Bb   = 16;          // batch
constexpr int SEQ  = 616;         // tokens
constexpr int CH   = 768;         // channels
constexpr int NH   = 12;          // heads
constexpr int HD   = 64;          // head dim
constexpr int HID  = 3072;        // mlp hidden
constexpr int MALL = Bb * SEQ;    // 9856 rows (divisible by 64)
constexpr int NP   = 640;         // padded seq for attention (40 tiles of 16)
constexpr int NTXT = 40;          // text tokens
constexpr int NIMG = SEQ - NTXT;  // 576 image tokens

// ---- helpers --------------------------------------------------------------
DEV unsigned short f2bf(float f) {                    // round-to-nearest-even
    unsigned int u = __builtin_bit_cast(unsigned int, f);
    u += 0x7FFFu + ((u >> 16) & 1u);
    return (unsigned short)(u >> 16);
}
DEV float bf2f(unsigned short h) {
    return __builtin_bit_cast(float, ((unsigned int)h) << 16);
}

// CDNA5 async DMA: copy 16 bytes global -> LDS, tracked by ASYNCcnt
DEV void async_copy16(unsigned lds_byte_addr, const void* gaddr) {
    asm volatile("global_load_async_to_lds_b128 %0, %1, off"
                 :: "v"(lds_byte_addr), "v"(gaddr) : "memory");
}
DEV void wait_async0() {
    asm volatile("s_wait_asynccnt 0" ::: "memory");
}

// B-fragment: lane takes 32 contiguous bytes (K 0..15 or 16..31 at its column)
DEV void ld_bfrag(Frag& f, const unsigned short* p) {
    const uint4* q = (const uint4*)p;
    uint4 a = q[0], b = q[1];
    f.d[0] = a.x; f.d[1] = a.y; f.d[2] = a.z; f.d[3] = a.w;
    f.d[4] = b.x; f.d[5] = b.y; f.d[6] = b.z; f.d[7] = b.w;
}
// A-fragment: lane-half 0 -> K {0..15}; half 1 -> K {16..31}
DEV void ld_afrag(Frag& f, const unsigned short* row, int half) {
    const uint4* q0 = (const uint4*)(row + half * 16);
    const uint4* q1 = (const uint4*)(row + half * 16 + 8);
    uint4 a = q0[0], b = q1[0];
    f.d[0] = a.x; f.d[1] = a.y; f.d[2] = a.z; f.d[3] = a.w;
    f.d[4] = b.x; f.d[5] = b.y; f.d[6] = b.z; f.d[7] = b.w;
}
DEV f32x8 wmma_bf16(const Frag& a, const Frag& b, f32x8 c) {
    return __builtin_amdgcn_wmma_f32_16x16x32_bf16(
        /*neg_a=*/false, a.v, /*neg_b=*/false, b.v,
        /*c_mod=*/(short)0, c, /*reuse_a=*/false, /*reuse_b=*/false);
}
// pack 16 f32 -> 16 bf16 into LDS (8 dword stores)
DEV void pack_store(unsigned short* dst, float4 w0, float4 w1, float4 w2, float4 w3) {
    unsigned int* bd = (unsigned int*)dst;
    bd[0] = (unsigned)f2bf(w0.x) | ((unsigned)f2bf(w0.y) << 16);
    bd[1] = (unsigned)f2bf(w0.z) | ((unsigned)f2bf(w0.w) << 16);
    bd[2] = (unsigned)f2bf(w1.x) | ((unsigned)f2bf(w1.y) << 16);
    bd[3] = (unsigned)f2bf(w1.z) | ((unsigned)f2bf(w1.w) << 16);
    bd[4] = (unsigned)f2bf(w2.x) | ((unsigned)f2bf(w2.y) << 16);
    bd[5] = (unsigned)f2bf(w2.z) | ((unsigned)f2bf(w2.w) << 16);
    bd[6] = (unsigned)f2bf(w3.x) | ((unsigned)f2bf(w3.y) << 16);
    bd[7] = (unsigned)f2bf(w3.z) | ((unsigned)f2bf(w3.w) << 16);
}

// ---------------------------------------------------------------------------
// qkv bias build: [q_bias | 0 | v_bias]
// ---------------------------------------------------------------------------
__global__ __launch_bounds__(256) void build_qkv_bias(
    const float* __restrict__ qb, const float* __restrict__ vb,
    float* __restrict__ out) {
    int i = blockIdx.x * 256 + threadIdx.x;
    if (i >= 3 * CH) return;
    float v = 0.f;
    if (i < CH) v = qb[i];
    else if (i >= 2 * CH) v = vb[i - 2 * CH];
    out[i] = v;
}

// ---------------------------------------------------------------------------
// LayerNorm (one 768-wide row per 256-thread block) -> bf16
// ---------------------------------------------------------------------------
__global__ __launch_bounds__(256) void ln1_kernel(
    const float* __restrict__ x, const float* __restrict__ g,
    const float* __restrict__ bb, unsigned short* __restrict__ out) {
    __shared__ float rs[8], rs2[8];
    __shared__ float smu, srstd;
    const int row = blockIdx.x, tid = threadIdx.x;
    const float* xr = x + (size_t)row * CH;
    float v0 = xr[tid], v1 = xr[tid + 256], v2 = xr[tid + 512];
    float s = v0 + v1 + v2, s2 = v0 * v0 + v1 * v1 + v2 * v2;
#pragma unroll
    for (int m = 16; m >= 1; m >>= 1) {
        s += __shfl_xor(s, m, 32);
        s2 += __shfl_xor(s2, m, 32);
    }
    if ((tid & 31) == 0) { rs[tid >> 5] = s; rs2[tid >> 5] = s2; }
    __syncthreads();
    if (tid == 0) {
        float S = 0.f, S2 = 0.f;
#pragma unroll
        for (int i = 0; i < 8; ++i) { S += rs[i]; S2 += rs2[i]; }
        float mu  = S * (1.0f / CH);
        float var = S2 * (1.0f / CH) - mu * mu;
        smu = mu; srstd = rsqrtf(var + 1e-5f);
    }
    __syncthreads();
    float mu = smu, rstd = srstd;
    unsigned short* op = out + (size_t)row * CH;
    op[tid]       = f2bf((v0 - mu) * rstd * g[tid]       + bb[tid]);
    op[tid + 256] = f2bf((v1 - mu) * rstd * g[tid + 256] + bb[tid + 256]);
    op[tid + 512] = f2bf((v2 - mu) * rstd * g[tid + 512] + bb[tid + 512]);
}

// LN2: text rows -> compacted text buffer, image rows -> compacted image buffer
__global__ __launch_bounds__(256) void ln2_kernel(
    const float* __restrict__ x1,
    const float* __restrict__ gt, const float* __restrict__ bt,
    const float* __restrict__ gi, const float* __restrict__ bi,
    unsigned short* __restrict__ out_t, unsigned short* __restrict__ out_i) {
    __shared__ float rs[8], rs2[8];
    __shared__ float smu, srstd;
    const int row = blockIdx.x, tid = threadIdx.x;
    const int b = row / SEQ, n = row % SEQ;
    const float* xr = x1 + (size_t)row * CH;
    float v0 = xr[tid], v1 = xr[tid + 256], v2 = xr[tid + 512];
    float s = v0 + v1 + v2, s2 = v0 * v0 + v1 * v1 + v2 * v2;
#pragma unroll
    for (int m = 16; m >= 1; m >>= 1) {
        s += __shfl_xor(s, m, 32);
        s2 += __shfl_xor(s2, m, 32);
    }
    if ((tid & 31) == 0) { rs[tid >> 5] = s; rs2[tid >> 5] = s2; }
    __syncthreads();
    if (tid == 0) {
        float S = 0.f, S2 = 0.f;
#pragma unroll
        for (int i = 0; i < 8; ++i) { S += rs[i]; S2 += rs2[i]; }
        float mu  = S * (1.0f / CH);
        float var = S2 * (1.0f / CH) - mu * mu;
        smu = mu; srstd = rsqrtf(var + 1e-5f);
    }
    __syncthreads();
    float mu = smu, rstd = srstd;
    const float* g; const float* bb; unsigned short* op;
    if (n < NTXT) { g = gt; bb = bt; op = out_t + (size_t)(b * NTXT + n) * CH; }
    else          { g = gi; bb = bi; op = out_i + (size_t)(b * NIMG + (n - NTXT)) * CH; }
    op[tid]       = f2bf((v0 - mu) * rstd * g[tid]       + bb[tid]);
    op[tid + 256] = f2bf((v1 - mu) * rstd * g[tid + 256] + bb[tid + 256]);
    op[tid + 512] = f2bf((v2 - mu) * rstd * g[tid + 512] + bb[tid + 512]);
}

// ---------------------------------------------------------------------------
// Generic bf16 WMMA GEMM:  out[m,o] = epi( sum_k A[m,k]*W[o,k] + bias[o] )
//   A: bf16 [M,K] row-major (staged with GLOBAL_LOAD_ASYNC_TO_LDS_B128)
//   W: f32 [O,K] row-major (converted to bf16 while staging)
//   64x64 block, 4 waves, double-buffered LDS, K step 32
// ---------------------------------------------------------------------------
enum { EPI_BF16_BIAS = 0, EPI_RESID1 = 1, EPI_GELU = 2, EPI_RESID2 = 3 };

template <int EPI>
__global__ __launch_bounds__(128) void gemm_bf16(
    const unsigned short* __restrict__ A, const float* __restrict__ W,
    const float* __restrict__ bias, const float* __restrict__ res,
    const float* __restrict__ gvec, unsigned short* __restrict__ outb,
    float* __restrict__ outf, int M, int O, int K, int RPB, int noff) {
    __shared__ __align__(16) unsigned short As[2][64][32];
    __shared__ __align__(16) unsigned short Bs[2][64][32];
    const int mB   = blockIdx.x * 64;
    const int oB   = blockIdx.y * 64;
    const int tid  = threadIdx.x;
    const int wave = tid >> 5, lane = tid & 31;
    const int half = lane >> 4, li = lane & 15;
    const int srow = tid >> 1, schk = (tid & 1) * 16;   // staging assignment

    f32x8 acc[4] = {};
    const unsigned short* Arow = A + (size_t)(mB + srow) * K + schk;
    const float*          Wrow = W + (size_t)(oB + srow) * K + schk;
    const int nk = K / 32;

    {   // prologue: stage tile 0 (A via async DMA, W via convert)
        unsigned dst = (unsigned)(size_t)&As[0][srow][schk];
        async_copy16(dst,      Arow);
        async_copy16(dst + 16, Arow + 8);
        const float4* wp = (const float4*)Wrow;
        pack_store(&Bs[0][srow][schk], wp[0], wp[1], wp[2], wp[3]);
        wait_async0();
    }
    __syncthreads();

    for (int it = 0; it < nk; ++it) {
        const int cur = it & 1, nxt = cur ^ 1;
        const bool more = (it + 1 < nk);
        float4 w0{}, w1{}, w2{}, w3{};
        if (more) {   // kick off next tile: async A DMA + W loads in flight
            unsigned dst = (unsigned)(size_t)&As[nxt][srow][schk];
            const unsigned short* src = Arow + (it + 1) * 32;
            async_copy16(dst,      src);
            async_copy16(dst + 16, src + 8);
            const float4* wp = (const float4*)(Wrow + (it + 1) * 32);
            w0 = wp[0]; w1 = wp[1]; w2 = wp[2]; w3 = wp[3];
        }
        // compute current tile: wave owns 16 M-rows x 64 N-cols (4 WMMA)
        Frag af;
        ld_afrag(af, &As[cur][wave * 16 + li][0], half);
#pragma unroll
        for (int j = 0; j < 4; ++j) {
            Frag bf;
            ld_bfrag(bf, &Bs[cur][j * 16 + li][half * 16]);
            acc[j] = wmma_bf16(af, bf, acc[j]);
        }
        if (more) {
            pack_store(&Bs[nxt][srow][schk], w0, w1, w2, w3);
            wait_async0();          // As[nxt] fully written by DMA
        }
        __syncthreads();            // publish next buffer to all waves
    }
    // ---- epilogue (C/D layout: row = half*8+i, col = li) ----
#pragma unroll
    for (int j = 0; j < 4; ++j) {
        const int ocol = oB + j * 16 + li;
        const float bv = bias[ocol];
#pragma unroll
        for (int i = 0; i < 8; ++i) {
            const int m = mB + wave * 16 + half * 8 + i;
            float v = acc[j][i] + bv;
            if constexpr (EPI == EPI_BF16_BIAS) {
                outb[(size_t)m * O + ocol] = f2bf(v);
            } else if constexpr (EPI == EPI_RESID1) {
                size_t idx = (size_t)m * O + ocol;
                outf[idx] = res[idx] + gvec[ocol] * v;
            } else if constexpr (EPI == EPI_GELU) {
                float gl = 0.5f * v * (1.0f + erff(v * 0.70710678118654752f));
                outb[(size_t)m * O + ocol] = f2bf(gl);
            } else {  // EPI_RESID2: compacted row -> original token
                int tok = (m / RPB) * SEQ + noff + (m % RPB);
                size_t idx = (size_t)tok * CH + ocol;
                outf[idx] = res[idx] + gvec[ocol] * v;
            }
        }
    }
}

// ---------------------------------------------------------------------------
// split qkv [M,2304] bf16 -> per-head q (scaled 1/8), k [BH,NP,64], vT [BH,64,NP]
// pads n in [616,640) with zeros
// ---------------------------------------------------------------------------
__global__ __launch_bounds__(256) void split_qkv(
    const unsigned short* __restrict__ qkv,
    unsigned short* __restrict__ q, unsigned short* __restrict__ k,
    unsigned short* __restrict__ vT) {
    const size_t idx = (size_t)blockIdx.x * 256 + threadIdx.x;
    const int d  = (int)(idx & (HD - 1));
    const int n  = (int)((idx >> 6) % NP);
    const int bh = (int)(idx / ((size_t)NP * HD));
    const int b  = bh / NH, h = bh % NH;
    unsigned short qv = 0, kv = 0, vv = 0;
    if (n < SEQ) {
        size_t base = (size_t)(b * SEQ + n) * (3 * CH) + h * HD + d;
        qv = f2bf(bf2f(qkv[base]) * 0.125f);   // scale = hd^-0.5 = 1/8 exactly
        kv = qkv[base + CH];
        vv = qkv[base + 2 * CH];
    }
    q[((size_t)bh * NP + n) * HD + d]  = qv;
    k[((size_t)bh * NP + n) * HD + d]  = kv;
    vT[((size_t)bh * HD + d) * NP + n] = vv;
}

// ---------------------------------------------------------------------------
// Flash attention: one (b,h,64-query block) per 128-thread workgroup.
// Wave owns a 16-row Q tile; loops 20 x (32 key columns).
// ---------------------------------------------------------------------------
__global__ __launch_bounds__(128) void flash_attn(
    const unsigned short* __restrict__ q, const unsigned short* __restrict__ k,
    const unsigned short* __restrict__ vT, const float* __restrict__ rel_bias,
    unsigned short* __restrict__ o) {
    __shared__ __align__(16) unsigned short Ps[4][16][32];
    const int bh   = blockIdx.x / (NP / 64);
    const int qblk = blockIdx.x % (NP / 64);
    const int b = bh / NH, h = bh % NH;
    const int wave = threadIdx.x >> 5, lane = threadIdx.x & 31;
    const int half = lane >> 4, li = lane & 15;
    const int qb = qblk * 64 + wave * 16;

    const unsigned short* qp = q  + (size_t)bh * NP * HD;
    const unsigned short* kp = k  + (size_t)bh * NP * HD;
    const unsigned short* vp = vT + (size_t)bh * HD * NP;

    Frag qa0, qa1;   // Q A-fragments, K(head-dim) 0..31 and 32..63
    {
        const unsigned short* row = qp + (size_t)(qb + li) * HD;
        ld_afrag(qa0, row, half);
        ld_afrag(qa1, row + 32, half);
    }
    f32x8 oacc[4] = {};
    float m_run[8], l_run[8];
#pragma unroll
    for (int i = 0; i < 8; ++i) { m_run[i] = -INFINITY; l_run[i] = 0.f; }

    for (int it = 0; it < NP / 32; ++it) {
        f32x8 s0 = {}, s1 = {};
        {   // S = Q . K^T for two 16-col sub-tiles (4 WMMA)
            const int kc0 = it * 32;
            Frag kb;
            const unsigned short* r0 = kp + (size_t)(kc0 + li) * HD + half * 16;
            ld_bfrag(kb, r0);       s0 = wmma_bf16(qa0, kb, s0);
            ld_bfrag(kb, r0 + 32);  s0 = wmma_bf16(qa1, kb, s0);
            const unsigned short* r1 = kp + (size_t)(kc0 + 16 + li) * HD + half * 16;
            ld_bfrag(kb, r1);       s1 = wmma_bf16(qa0, kb, s1);
            ld_bfrag(kb, r1 + 32);  s1 = wmma_bf16(qa1, kb, s1);
        }
        const int qrow0 = qb + half * 8;
        const int kc = it * 32 + li;
        const int kcc0 = (kc < SEQ) ? kc : (SEQ - 1);          // clamped, branchless
        const int kcc1 = (kc + 16 < SEQ) ? kc + 16 : (SEQ - 1);
#pragma unroll
        for (int i = 0; i < 8; ++i) {
            const int qrow = qrow0 + i;
            const int qr = (qrow < SEQ) ? qrow : (SEQ - 1);
            const float* rb = rel_bias + ((size_t)h * SEQ + qr) * SEQ;
            float b0 = rb[kcc0], b1 = rb[kcc1];                // unconditional loads
            float a = s0[i], c = s1[i];
            a = (kc < SEQ)      ? ((qrow < SEQ) ? a + b0 : a) : -1e30f;
            c = (kc + 16 < SEQ) ? ((qrow < SEQ) ? c + b1 : c) : -1e30f;
            // online softmax: per-row reduce over the 16 lanes of this half
            float mx = fmaxf(a, c);
#pragma unroll
            for (int msk = 8; msk >= 1; msk >>= 1) mx = fmaxf(mx, __shfl_xor(mx, msk, 32));
            float mnew = fmaxf(m_run[i], mx);
            float corr = __expf(m_run[i] - mnew);
            float p0 = __expf(a - mnew), p1 = __expf(c - mnew);
            float rs = p0 + p1;
#pragma unroll
            for (int msk = 8; msk >= 1; msk >>= 1) rs += __shfl_xor(rs, msk, 32);
            l_run[i] = l_run[i] * corr + rs;
            m_run[i] = mnew;
            oacc[0][i] = oacc[0][i] * corr;
            oacc[1][i] = oacc[1][i] * corr;
            oacc[2][i] = oacc[2][i] * corr;
            oacc[3][i] = oacc[3][i] * corr;
            // C/D layout -> LDS so we can re-read as A fragment
            Ps[wave][half * 8 + i][li]      = f2bf(p0);
            Ps[wave][half * 8 + i][16 + li] = f2bf(p1);
        }
        __syncthreads();
        Frag pa;
        ld_afrag(pa, &Ps[wave][li][0], half);
#pragma unroll
        for (int j = 0; j < 4; ++j) {    // O += P . V (vT rows contiguous in k)
            Frag vb;
            ld_bfrag(vb, vp + (size_t)(j * 16 + li) * NP + it * 32 + half * 16);
            oacc[j] = wmma_bf16(pa, vb, oacc[j]);
        }
        __syncthreads();
    }
    // normalize + store [B*SEQ, CH] bf16 (head-packed columns)
#pragma unroll
    for (int i = 0; i < 8; ++i) {
        const int qrow = qb + half * 8 + i;
        if (qrow >= SEQ) continue;
        const float inv = 1.0f / l_run[i];
        const size_t base = (size_t)(b * SEQ + qrow) * CH + h * HD;
#pragma unroll
        for (int j = 0; j < 4; ++j)
            o[base + j * 16 + li] = f2bf(oacc[j][i] * inv);
    }
}

// ---------------------------------------------------------------------------
// host-side orchestration
// ---------------------------------------------------------------------------
extern "C" void kernel_launch(void* const* d_in, const int* in_sizes, int n_in,
                              void* d_out, int out_size, void* d_ws, size_t ws_size,
                              hipStream_t stream) {
    (void)in_sizes; (void)n_in; (void)out_size; (void)ws_size;
    const float* x      = (const float*)d_in[0];
    const float* relb   = (const float*)d_in[1];
    const float* ln1_g  = (const float*)d_in[2];
    const float* ln1_b  = (const float*)d_in[3];
    const float* w_qkv  = (const float*)d_in[4];
    const float* q_bias = (const float*)d_in[5];
    const float* v_bias = (const float*)d_in[6];
    const float* w_proj = (const float*)d_in[7];
    const float* b_proj = (const float*)d_in[8];
    const float* gamma1 = (const float*)d_in[9];
    const float* gamma2 = (const float*)d_in[10];
    const float* ln2t_g = (const float*)d_in[11];
    const float* ln2t_b = (const float*)d_in[12];
    const float* ln2i_g = (const float*)d_in[13];
    const float* ln2i_b = (const float*)d_in[14];
    const float* wt1    = (const float*)d_in[15];
    const float* bt1    = (const float*)d_in[16];
    const float* wt2    = (const float*)d_in[17];
    const float* bt2    = (const float*)d_in[18];
    const float* wi1    = (const float*)d_in[19];
    const float* bi1    = (const float*)d_in[20];
    const float* wi2    = (const float*)d_in[21];
    const float* bi2    = (const float*)d_in[22];
    float* out = (float*)d_out;

    // workspace layout (regions reused across phases; ~153 MB total)
    const size_t SZ_NLN = (size_t)MALL * CH * 2;            // ln1 out; later ln2 t+i
    const size_t SZ_X1  = (size_t)MALL * CH * 4;            // residual fp32
    const size_t SZ_O   = (size_t)MALL * CH * 2;            // attention out bf16
    const size_t SZ_K   = (size_t)Bb * NH * NP * HD * 2;    // K bf16 (padded)
    const size_t SZ_VT  = SZ_K;                             // V^T bf16
    const size_t SZ_QKV = (size_t)MALL * 3 * CH * 2;        // qkv bf16
    const size_t SZ_BIG = SZ_QKV + SZ_K;                    // qkv+q, later t_h+i_h
    char* p = (char*)d_ws;
    unsigned short* nln   = (unsigned short*)p; p += SZ_NLN;
    float*          x1    = (float*)p;          p += SZ_X1;
    unsigned short* obuf  = (unsigned short*)p; p += SZ_O;
    unsigned short* kbuf  = (unsigned short*)p; p += SZ_K;
    unsigned short* vtbuf = (unsigned short*)p; p += SZ_VT;
    char*           big   = p;                  p += SZ_BIG;
    float*          qkvb  = (float*)p;          p += 3 * CH * 4;

    unsigned short* qkvbuf = (unsigned short*)big;
    unsigned short* qbuf   = (unsigned short*)(big + SZ_QKV);
    unsigned short* tln = nln;                                   // reuse nln
    unsigned short* iln = nln + (size_t)Bb * NTXT * CH;
    unsigned short* th  = (unsigned short*)big;                  // reuse big
    unsigned short* ih  = (unsigned short*)(big + (size_t)Bb * NTXT * HID * 2);

    build_qkv_bias<<<(3 * CH + 255) / 256, 256, 0, stream>>>(q_bias, v_bias, qkvb);
    ln1_kernel<<<MALL, 256, 0, stream>>>(x, ln1_g, ln1_b, nln);
    gemm_bf16<EPI_BF16_BIAS><<<dim3(MALL / 64, 3 * CH / 64), 128, 0, stream>>>(
        nln, w_qkv, qkvb, nullptr, nullptr, qkvbuf, nullptr, MALL, 3 * CH, CH, 1, 0);
    split_qkv<<<(Bb * NH * NP * HD) / 256, 256, 0, stream>>>(qkvbuf, qbuf, kbuf, vtbuf);
    flash_attn<<<Bb * NH * (NP / 64), 128, 0, stream>>>(qbuf, kbuf, vtbuf, relb, obuf);
    gemm_bf16<EPI_RESID1><<<dim3(MALL / 64, CH / 64), 128, 0, stream>>>(
        obuf, w_proj, b_proj, x, gamma1, nullptr, x1, MALL, CH, CH, 1, 0);
    ln2_kernel<<<MALL, 256, 0, stream>>>(x1, ln2t_g, ln2t_b, ln2i_g, ln2i_b, tln, iln);
    gemm_bf16<EPI_GELU><<<dim3(Bb * NTXT / 64, HID / 64), 128, 0, stream>>>(
        tln, wt1, bt1, nullptr, nullptr, th, nullptr, Bb * NTXT, HID, CH, 1, 0);
    gemm_bf16<EPI_GELU><<<dim3(Bb * NIMG / 64, HID / 64), 128, 0, stream>>>(
        iln, wi1, bi1, nullptr, nullptr, ih, nullptr, Bb * NIMG, HID, CH, 1, 0);
    gemm_bf16<EPI_RESID2><<<dim3(Bb * NTXT / 64, CH / 64), 128, 0, stream>>>(
        th, wt2, bt2, x1, gamma2, nullptr, out, Bb * NTXT, CH, HID, NTXT, 0);
    gemm_bf16<EPI_RESID2><<<dim3(Bb * NIMG / 64, CH / 64), 128, 0, stream>>>(
        ih, wi2, bi2, x1, gamma2, nullptr, out, Bb * NIMG, CH, HID, NIMG, NTXT);
}
